// D1Layer_32246614458525
// MI455X (gfx1250) — matrix operand order repaired and checked by reading, not compiled, not run on hardware
//
#include <hip/hip_runtime.h>

typedef __attribute__((ext_vector_type(16))) __bf16 v16bf;
typedef __attribute__((ext_vector_type(8)))  float  v8f;

// ---------------------------------------------------------------------------
// Quantization: for each flat element of x, argmin over 256 codes of
// ||e_k||^2 - 2 * (powers(v) . e_k)   (the sum-term is constant per element)
// Output index permuted per reference: q[b,d] = ind[d*1024 + b], stored bf16.
// ---------------------------------------------------------------------------
__global__ __launch_bounds__(256) void quantize_kernel(
    const float* __restrict__ x, const float* __restrict__ embW,
    __bf16* __restrict__ q)
{
  __shared__ float sE[256][8];
  __shared__ float sN[256];
  const int t = threadIdx.x;
  {
    float4 e0 = *(const float4*)(embW + t * 8);
    float4 e1 = *(const float4*)(embW + t * 8 + 4);
    sE[t][0] = e0.x; sE[t][1] = e0.y; sE[t][2] = e0.z; sE[t][3] = e0.w;
    sE[t][4] = e1.x; sE[t][5] = e1.y; sE[t][6] = e1.z; sE[t][7] = e1.w;
    sN[t] = e0.x*e0.x + e0.y*e0.y + e0.z*e0.z + e0.w*e0.w +
            e1.x*e1.x + e1.y*e1.y + e1.z*e1.z + e1.w*e1.w;
  }
  __syncthreads();

  const int i = blockIdx.x * 256 + t;      // flat source index (coalesced read)
  const float v = x[i];
  float p[8];
  p[0] = v;
#pragma unroll
  for (int j = 1; j < 8; ++j) p[j] = p[j - 1] * v;

  float best = 3.402823466e38f;
  int bi = 0;
  for (int k = 0; k < 256; ++k) {
    float dot = 0.f;
#pragma unroll
    for (int j = 0; j < 8; ++j) dot += p[j] * sE[k][j];
    const float dist = sN[k] - 2.f * dot;
    if (dist < best) { best = dist; bi = k; }   // strict < -> first-min, matches argmin
  }
  // i = d*1024 + b  ->  out position b*512 + d
  const int o = (i & 1023) * 512 + (i >> 10);
  q[o] = (__bf16)(float)bi;                     // <=255, exact in bf16
}

// ---------------------------------------------------------------------------
// C = relu(A[M,K](bf16) @ W[N,K](f32)^T + bias[N]); out bf16 or f32.
// Block tile 128x128, 8 waves, wave tile 32x64 (2x4 v_wmma_f32_16x16x32_bf16).
// sA  : [128][40]  m-major (pad 40 -> 80B rows, 16B aligned, spread banks)
// sBT : [32][136]  k-major (pad 136 -> 272B rows; lane-l row gather hits
//                  distinct bank pairs; B fragment = 16 per-element LDS reads)
// Global tile for step kk+32 is staged in registers while step kk computes.
// ---------------------------------------------------------------------------
template <bool OUT_F32>
__global__ __launch_bounds__(256) void gemm_relu_kernel(
    const __bf16* __restrict__ A,
    const float*  __restrict__ W,
    const float*  __restrict__ bias,
    __bf16* __restrict__ outB,
    float*  __restrict__ outF,
    int M, int N, int K)
{
  __shared__ alignas(16) __bf16 sA[128][40];
  __shared__ alignas(16) __bf16 sBT[32][136];

  const int t     = threadIdx.x;
  const int lane  = t & 31;
  const int wave  = t >> 5;        // 0..7
  const int waveM = wave >> 1;     // 0..3  (32 rows each)
  const int waveN = wave & 1;      // 0..1  (64 cols each)
  const int m0 = blockIdx.y * 128;
  const int n0 = blockIdx.x * 128;

  // A-fill mapping: 16 contiguous bf16 per thread
  const int fr = t >> 1;           // row 0..127
  const int fh = (t & 1) * 16;     // col half
  // W-fill mapping: 4(n) x 4(k) subtile per thread
  const int nq = t & 31;           // n = 4*nq
  const int kq = t >> 5;           // k = 4*kq  (kq 0..7 covers k 0..31)

  v8f acc[2][4];
#pragma unroll
  for (int mi = 0; mi < 2; ++mi)
#pragma unroll
    for (int ni = 0; ni < 4; ++ni)
#pragma unroll
      for (int r = 0; r < 8; ++r) acc[mi][ni][r] = 0.f;

  const int mrow  = lane & 15;
  const int khalf = (lane >> 4) * 8;

  // ---- register-staged global tile (double buffer vs LDS) ----
  uint4 ra0, ra1;
  float rw[4][4];
  auto load_tile = [&](int kk) {
    const __bf16* sa = A + (size_t)(m0 + fr) * K + kk + fh;
    ra0 = *(const uint4*)(sa);
    ra1 = *(const uint4*)(sa + 8);
    const float* wp = W + (size_t)(n0 + 4 * nq) * K + kk + 4 * kq;
#pragma unroll
    for (int i = 0; i < 4; ++i) {
      float4 f = *(const float4*)(wp + (size_t)i * K);
      rw[i][0] = f.x; rw[i][1] = f.y; rw[i][2] = f.z; rw[i][3] = f.w;
    }
    if (kk + 32 < K) {
      __builtin_prefetch(sa + 32, 0, 0);            // global_prefetch_b8
      __builtin_prefetch(wp + 32, 0, 0);
    }
  };

  load_tile(0);

  for (int kk = 0; kk < K; kk += 32) {
    __syncthreads();
    // ---- commit staged tile to LDS ----
    *(uint4*)&sA[fr][fh]     = ra0;
    *(uint4*)&sA[fr][fh + 8] = ra1;
#pragma unroll
    for (int j = 0; j < 4; ++j) {     // transpose 4x4: store 4 bf16 along n
      union { __bf16 h[4]; uint2 u; } p;
      p.h[0] = (__bf16)rw[0][j];
      p.h[1] = (__bf16)rw[1][j];
      p.h[2] = (__bf16)rw[2][j];
      p.h[3] = (__bf16)rw[3][j];
      *(uint2*)&sBT[4 * kq + j][4 * nq] = p.u;
    }
    __syncthreads();

    // ---- stage next global tile (overlaps with compute below) ----
    if (kk + 32 < K) load_tile(kk + 32);

    // ---- A fragments: two contiguous 16B LDS loads per fragment ----
    union { v16bf v; uint4 u[2]; } afrag[2];
#pragma unroll
    for (int mi = 0; mi < 2; ++mi) {
      const int m = waveM * 32 + mi * 16 + mrow;
      afrag[mi].u[0] = *(const uint4*)&sA[m][khalf];        // K 0..15 half
      afrag[mi].u[1] = *(const uint4*)&sA[m][16 + khalf];   // K 16..31 half
    }
    // ---- B fragments: lane = N-col, 16 K elems in same order as A ----
    union { v16bf v; uint4 u[2]; } bfrag[4];
#pragma unroll
    for (int ni = 0; ni < 4; ++ni) {
      const int nb = waveN * 64 + ni * 16;
#pragma unroll
      for (int s = 0; s < 16; ++s)
        bfrag[ni].v[s] = sBT[s < 8 ? khalf + s : 16 + khalf + (s - 8)][nb + mrow];
    }

#pragma unroll
    for (int mi = 0; mi < 2; ++mi)
#pragma unroll
      for (int ni = 0; ni < 4; ++ni)
        acc[mi][ni] = __builtin_amdgcn_wmma_f32_16x16x32_bf16(
            false, afrag[mi].v, false, bfrag[ni].v, (short)0, acc[mi][ni],
            false, false);
  }

  // Epilogue: bias + ReLU, C/D layout: VGPR r -> M=r (lanes 0-15) / M=8+r (16-31)
  const int mbase = m0 + waveM * 32 + ((lane >> 4) ? 8 : 0);
  const int ncol  = lane & 15;
#pragma unroll
  for (int mi = 0; mi < 2; ++mi) {
#pragma unroll
    for (int ni = 0; ni < 4; ++ni) {
      const int n  = n0 + waveN * 64 + ni * 16 + ncol;
      const float bv = bias[n];
#pragma unroll
      for (int r = 0; r < 8; ++r) {
        const int m = mbase + mi * 16 + r;
        float val = acc[mi][ni][r] + bv;
        val = val > 0.f ? val : 0.f;
        if (OUT_F32) outF[(size_t)m * N + n] = val;
        else         outB[(size_t)m * N + n] = (__bf16)val;
      }
    }
  }
}

__global__ void write_loss_kernel(float* p) {
  if (threadIdx.x == 0) *p = 0.0f;
}

// ---------------------------------------------------------------------------
extern "C" void kernel_launch(void* const* d_in, const int* in_sizes, int n_in,
                              void* d_out, int out_size, void* d_ws, size_t ws_size,
                              hipStream_t stream) {
  (void)in_sizes; (void)n_in; (void)ws_size;
  const float* x     = (const float*)d_in[0];
  const float* embW  = (const float*)d_in[1];
  const float* W_in  = (const float*)d_in[2];
  const float* b_in  = (const float*)d_in[3];
  const float* W_h   = (const float*)d_in[4];
  const float* b_h   = (const float*)d_in[5];
  const float* W_out = (const float*)d_in[6];
  const float* b_out = (const float*)d_in[7];
  float* out = (float*)d_out;

  char* ws = (char*)d_ws;
  __bf16* q  = (__bf16*)ws;                                        // 1024*512
  __bf16* hA = (__bf16*)(ws + (size_t)1024 * 512 * 2);             // 1024*2048
  __bf16* hB = (__bf16*)(ws + (size_t)1024 * 512 * 2
                            + (size_t)1024 * 2048 * 2);            // 1024*2048

  // quantize: 524288 elements
  quantize_kernel<<<2048, 256, 0, stream>>>(x, embW, q);

  // input layer: [1024,512] @ [512,2048] -> [1024,2048]
  gemm_relu_kernel<false><<<dim3(16, 8), 256, 0, stream>>>(
      q, W_in, b_in, hA, nullptr, 1024, 2048, 512);

  // 4 shared hidden layers: [1024,2048] @ [2048,2048]
  gemm_relu_kernel<false><<<dim3(16, 8), 256, 0, stream>>>(
      hA, W_h, b_h, hB, nullptr, 1024, 2048, 2048);
  gemm_relu_kernel<false><<<dim3(16, 8), 256, 0, stream>>>(
      hB, W_h, b_h, hA, nullptr, 1024, 2048, 2048);
  gemm_relu_kernel<false><<<dim3(16, 8), 256, 0, stream>>>(
      hA, W_h, b_h, hB, nullptr, 1024, 2048, 2048);
  gemm_relu_kernel<false><<<dim3(16, 8), 256, 0, stream>>>(
      hB, W_h, b_h, hA, nullptr, 1024, 2048, 2048);

  // output layer: [1024,2048] @ [2048,512] -> [1024,512], fp32 to d_out
  gemm_relu_kernel<true><<<dim3(4, 8), 256, 0, stream>>>(
      hA, W_out, b_out, nullptr, out, 1024, 512, 2048);

  // q_latent_loss scalar (0.0 in eval mode) appended after f
  write_loss_kernel<<<1, 1, 0, stream>>>(out + (out_size - 1));
}
